// RnnSenderReinforce_51324859187315
// MI455X (gfx1250) — compile-verified, matrix operand order checked by reading, not run on hardware
//
#include <hip/hip_runtime.h>
#include <hip/hip_bf16.h>
#include <math.h>

// ---------------------------------------------------------------------------
// Problem dims
// ---------------------------------------------------------------------------
#define B_TOT   8192
#define IN_DIM  512
#define H_DIM   512
#define E_DIM   256
#define V_DIM   1024
#define NSTEPS  31
#define MAX_LEN 32

#define M_TILE   64                    // batch rows per block (4 WMMA row-tiles)
#define NBLOCKS  (B_TOT / M_TILE)      // 128
#define NTHREADS 256
#define NW       8                     // waves per block (wave32)
#define RT       4                     // row-tiles register-blocked per wave

// LDS row strides (in u16 elements), padded to kill bank conflicts, 16B-aligned
#define HS_LD   520
#define INP_LD  264
#define RZ_LD   1040
#define NN_LD   520
#define LG_LD   1040

// dynamic LDS layout (bytes)
#define HS_OFF   0
#define INP_OFF  (M_TILE * HS_LD * 2)                       // 66560
#define POOL_OFF (INP_OFF + M_TILE * INP_LD * 2)            // 100352
#define NB_OFF   (POOL_OFF + M_TILE * RZ_LD * 2)            // 233472
#define LDS_BYTES (NB_OFF + M_TILE * NN_LD * 2)             // 300032

// workspace layout (bytes)
#define X_BF_OFF    0u
#define AGW_OFF     8388608u
#define WIH_OFF     8912896u
#define WHH_OFF     9699328u
#define OUTW_OFF    11272192u
#define EMB_OFF     12320768u

typedef unsigned short u16;
typedef __bf16        bf16x16 __attribute__((ext_vector_type(16)));
typedef float         f32x8   __attribute__((ext_vector_type(8)));
typedef unsigned int  u32x4   __attribute__((ext_vector_type(4)));

union Frag16 {
    u32x4   q[2];
    bf16x16 v;
};

__device__ __forceinline__ u16 f2bf(float f) {
    unsigned b = __float_as_uint(f);
    unsigned r = b + 0x7FFFu + ((b >> 16) & 1u);   // round-to-nearest-even
    return (u16)(r >> 16);
}
__device__ __forceinline__ float bf2f(u16 h) {
    return __uint_as_float(((unsigned)h) << 16);
}

__device__ __forceinline__ float wred_max(float v) {
    #pragma unroll
    for (int o = 16; o > 0; o >>= 1) v = fmaxf(v, __shfl_xor(v, o, 32));
    return v;
}
__device__ __forceinline__ float wred_sum(float v) {
    #pragma unroll
    for (int o = 16; o > 0; o >>= 1) v += __shfl_xor(v, o, 32);
    return v;
}

__device__ __forceinline__ unsigned hashmix(unsigned x) {
    x ^= x >> 17; x *= 0xed5ad4bbu;
    x ^= x >> 11; x *= 0xac4c1b51u;
    x ^= x >> 15; x *= 0x31848babu;
    x ^= x >> 14;
    return x;
}

// ---------------------------------------------------------------------------
// Register-blocked WMMA column strip:
//   acc[rt] += act[row0 + rt*16 .., K] @ W[n0.., K]^T      (bf16 -> f32)
// B (weight) fragment loaded ONCE per K-chunk, reused across 4 row-tiles ->
// 4 independent WMMA accumulation chains, 4x less L2 weight traffic.
// K loop deliberately NOT unrolled (#pragma unroll 1): with full unrolling
// LICM hoists the jn-invariant A fragments out of the caller's n-tile loop,
// blowing past 256 VGPRs and spilling to scratch (observed in round 2 asm).
// Fragment layouts per ISA 7.12.2: per-lane m/n = lane&15, K-half by lane>>4,
// two 16B chunks at k0+sel*8 and k0+16+sel*8.
// ---------------------------------------------------------------------------
__device__ __forceinline__ void mma_col(f32x8 (&acc)[RT],
                                        const u16* act, int actLd, int row0,
                                        const u16* w, int wldK, int n0,
                                        int kTot, int lane)
{
    const int sel8 = (lane >> 4) << 3;            // 0 or 8 elements
    const u16* arow = act + (size_t)(row0 + (lane & 15)) * actLd + sel8;
    const u16* brow = w + (size_t)(n0 + (lane & 15)) * wldK + sel8;
    const size_t rtStride = (size_t)16 * actLd;

    #pragma unroll 1
    for (int k0 = 0; k0 < kTot; k0 += 32) {
        Frag16 b;
        b.q[0] = *reinterpret_cast<const u32x4*>(brow + k0);
        b.q[1] = *reinterpret_cast<const u32x4*>(brow + k0 + 16);
        __builtin_prefetch(brow + k0 + 32, 0, 1);  // global_prefetch_b8
        #pragma unroll
        for (int rt = 0; rt < RT; ++rt) {
            Frag16 a;
            const u16* ar = arow + rt * rtStride;
            a.q[0] = *reinterpret_cast<const u32x4*>(ar + k0);
            a.q[1] = *reinterpret_cast<const u32x4*>(ar + k0 + 16);
            acc[rt] = __builtin_amdgcn_wmma_f32_16x16x32_bf16(
                          false, a.v, false, b.v, (short)0, acc[rt], false, false);
        }
    }
}

// ---------------------------------------------------------------------------
// fp32 -> bf16 conversion (weights + x), grid-stride
// ---------------------------------------------------------------------------
__global__ void f2bf_kernel(const float* __restrict__ src, u16* __restrict__ dst, int n) {
    for (int i = blockIdx.x * blockDim.x + threadIdx.x; i < n;
         i += gridDim.x * blockDim.x)
        dst[i] = f2bf(src[i]);
}

// ---------------------------------------------------------------------------
// Persistent GRU rollout kernel.  One block owns M_TILE=64 batch rows; h and
// inp live in LDS across all 31 steps.  All GEMMs via v_wmma_f32_16x16x32_bf16
// with 4-way M register blocking.
// ---------------------------------------------------------------------------
__global__ __launch_bounds__(NTHREADS, 1)
void rnn_sender_kernel(const u16* __restrict__ x_bf,
                       const u16* __restrict__ agw_bf,   // [H][IN]
                       const float* __restrict__ agent_b,
                       const u16* __restrict__ wih_bf,   // [3H][E]
                       const u16* __restrict__ whh_bf,   // [3H][H]
                       const float* __restrict__ b_ih,
                       const float* __restrict__ b_hh,
                       const u16* __restrict__ outw_bf,  // [V][H]
                       const float* __restrict__ out_b,
                       const u16* __restrict__ emb_bf,   // [V][E]
                       const float* __restrict__ sos,    // [E]
                       float* __restrict__ out)          // seq | lps | ents
{
    extern __shared__ unsigned char smem[];
    u16* hs  = (u16*)(smem + HS_OFF);    // [64][HS_LD]  persistent h (bf16)
    u16* inp = (u16*)(smem + INP_OFF);   // [64][INP_LD] step input (bf16)
    u16* rz  = (u16*)(smem + POOL_OFF);  // [64][RZ_LD]  r | z  (cols 0..1023)
    u16* nb  = (u16*)(smem + NB_OFF);    // [64][NN_LD]  n  (also parks gh_n)
    u16* lg  = rz;                        // logits overlay [64][LG_LD]
    __shared__ int tokS[M_TILE];

    const int lane = threadIdx.x & 31;
    const int wave = threadIdx.x >> 5;
    const int n16  = lane & 15;
    const int sel  = lane >> 4;
    const int blk0 = blockIdx.x * M_TILE;

    float* seqO = out;
    float* lpsO = out + (size_t)B_TOT * MAX_LEN;
    float* entO = out + 2 * (size_t)B_TOT * MAX_LEN;

    // ---- prologue: inp0 = sos,  h0 = x @ agent_w^T + agent_b -------------
    for (int i = threadIdx.x; i < M_TILE * E_DIM; i += NTHREADS) {
        int m = i >> 8, j = i & (E_DIM - 1);
        inp[m * INP_LD + j] = f2bf(sos[j]);
    }
    for (int jn = wave; jn < H_DIM / 16; jn += NW) {     // 32 n-tiles
        int col0 = jn << 4;
        f32x8 acc[RT] = {};
        mma_col(acc, x_bf, IN_DIM, blk0, agw_bf, IN_DIM, col0, IN_DIM, lane);
        float bias = agent_b[col0 + n16];
        #pragma unroll
        for (int rt = 0; rt < RT; ++rt)
            #pragma unroll
            for (int r = 0; r < 8; ++r) {
                int m = (rt << 4) + r + (sel << 3);
                hs[m * HS_LD + col0 + n16] = f2bf(acc[rt][r] + bias);
            }
    }
    __syncthreads();

    // ---- 31 decode steps --------------------------------------------------
    for (int step = 0; step < NSTEPS; ++step) {
        // stage 1: r,z = sigmoid(inp@Wih[0:2H]^T + h@Whh[0:2H]^T + b)
        for (int jn = wave; jn < 64; jn += NW) {
            int w0 = jn << 4;                        // w0 == rz column base
            f32x8 acc[RT] = {};
            mma_col(acc, inp, INP_LD, 0, wih_bf, E_DIM, w0, E_DIM, lane);
            mma_col(acc, hs,  HS_LD,  0, whh_bf, H_DIM, w0, H_DIM, lane);
            float bias = b_ih[w0 + n16] + b_hh[w0 + n16];
            #pragma unroll
            for (int rt = 0; rt < RT; ++rt)
                #pragma unroll
                for (int r = 0; r < 8; ++r) {
                    int m = (rt << 4) + r + (sel << 3);
                    float v = 1.0f / (1.0f + __expf(-(acc[rt][r] + bias)));
                    rz[m * RZ_LD + w0 + n16] = f2bf(v);
                }
        }
        __syncthreads();

        // stage 2: n = tanh(gi_n + b_in + r * (gh_n + b_hn))
        // Two passes through the wave-private nb tile to keep only one 4-way
        // accumulator set live at a time (register pressure).
        for (int jn = wave; jn < 32; jn += NW) {
            int col0 = jn << 4;
            int w0 = 2 * H_DIM + col0;               // n-third rows of W
            float bi = b_ih[w0 + n16], bh = b_hh[w0 + n16];
            {
                f32x8 ah[RT] = {};
                mma_col(ah, hs, HS_LD, 0, whh_bf, H_DIM, w0, H_DIM, lane);
                #pragma unroll
                for (int rt = 0; rt < RT; ++rt)
                    #pragma unroll
                    for (int r = 0; r < 8; ++r) {
                        int m = (rt << 4) + r + (sel << 3);
                        nb[m * NN_LD + col0 + n16] = f2bf(ah[rt][r] + bh);
                    }
            }
            {
                f32x8 ai[RT] = {};
                mma_col(ai, inp, INP_LD, 0, wih_bf, E_DIM, w0, E_DIM, lane);
                #pragma unroll
                for (int rt = 0; rt < RT; ++rt)
                    #pragma unroll
                    for (int r = 0; r < 8; ++r) {
                        int m = (rt << 4) + r + (sel << 3);
                        float rv  = bf2f(rz[m * RZ_LD + col0 + n16]);
                        float ghv = bf2f(nb[m * NN_LD + col0 + n16]);
                        float v   = tanhf((ai[rt][r] + bi) + rv * ghv);
                        nb[m * NN_LD + col0 + n16] = f2bf(v);
                    }
            }
        }
        __syncthreads();

        // stage 3: h = (1-z)*n + z*h   (element-wise, in LDS)
        for (int i = threadIdx.x; i < M_TILE * H_DIM; i += NTHREADS) {
            int m = i >> 9, j = i & (H_DIM - 1);
            float z = bf2f(rz[m * RZ_LD + H_DIM + j]);
            float n = bf2f(nb[m * NN_LD + j]);
            float h = bf2f(hs[m * HS_LD + j]);
            hs[m * HS_LD + j] = f2bf((1.0f - z) * n + z * h);
        }
        __syncthreads();

        // stage 4: logits = h @ out_w^T + out_b  (overlays rz pool)
        for (int jn = wave; jn < V_DIM / 16; jn += NW) {
            int col0 = jn << 4;
            f32x8 acc[RT] = {};
            mma_col(acc, hs, HS_LD, 0, outw_bf, H_DIM, col0, H_DIM, lane);
            float bias = out_b[col0 + n16];
            #pragma unroll
            for (int rt = 0; rt < RT; ++rt)
                #pragma unroll
                for (int r = 0; r < 8; ++r) {
                    int m = (rt << 4) + r + (sel << 3);
                    lg[m * LG_LD + col0 + n16] = f2bf(acc[rt][r] + bias);
                }
        }
        __syncthreads();

        // stage 5: per-row log-softmax, entropy, categorical sample (wave32)
        for (int rr = 0; rr < M_TILE / NW; ++rr) {
            int m = wave * (M_TILE / NW) + rr;
            int grow = blk0 + m;
            const u16* lrow = lg + m * LG_LD;
            const int base = lane * 32;

            float la[32];
            float mx = -3.0e38f;
            #pragma unroll
            for (int j = 0; j < 32; ++j) {
                la[j] = bf2f(lrow[base + j]);
                mx = fmaxf(mx, la[j]);
            }
            mx = wred_max(mx);

            float ls = 0.0f;
            #pragma unroll
            for (int j = 0; j < 32; ++j) ls += __expf(la[j] - mx);
            float total = wred_sum(ls);
            float logZ  = mx + __logf(total);

            float es = 0.0f;
            #pragma unroll
            for (int j = 0; j < 32; ++j)
                es += __expf(la[j] - mx) * (la[j] - logZ);
            float ent = -wred_sum(es) / total;

            // CDF inversion: prefix-scan lane sums, locate crossing
            float u = (float)(hashmix(0x9E3779B9u * (unsigned)step +
                                      0x85EBCA6Bu * (unsigned)grow + 0x1234567u) >> 8)
                      * (1.0f / 16777216.0f);
            float target = u * total;
            float incl = ls;
            #pragma unroll
            for (int o = 1; o < 32; o <<= 1) {
                float t = __shfl_up(incl, o, 32);
                if (lane >= o) incl += t;
            }
            float excl = incl - ls;

            int tok = -1;
            if (target >= excl && target < incl) {
                float c = excl;
                #pragma unroll
                for (int j = 0; j < 32; ++j) {
                    c += __expf(la[j] - mx);
                    if (target < c) { tok = base + j; break; }
                }
                if (tok < 0) tok = base + 31;
            }
            unsigned long long bal = __ballot(tok >= 0);
            int srcLane = bal ? (int)(__ffsll(bal) - 1) : 31;
            int tokSel = __shfl(tok, srcLane, 32);
            if (tokSel < 0) tokSel = V_DIM - 1;

            float lp = bf2f(lrow[tokSel]) - logZ;
            if (lane == 0) {
                tokS[m] = tokSel;
                seqO[(size_t)grow * MAX_LEN + step] = (float)tokSel;
                lpsO[(size_t)grow * MAX_LEN + step] = lp;
                entO[(size_t)grow * MAX_LEN + step] = ent;
            }
        }
        __syncthreads();

        // stage 6: inp = emb[tok]  (L2-hot bf16 gather)
        for (int i = threadIdx.x; i < M_TILE * E_DIM; i += NTHREADS) {
            int m = i >> 8, j = i & (E_DIM - 1);
            inp[m * INP_LD + j] = emb_bf[(size_t)tokS[m] * E_DIM + j];
        }
        __syncthreads();
    }

    // force_eos column: zeros in slot 31 for seq / lps / ents
    if (threadIdx.x < M_TILE) {
        size_t grow = blk0 + threadIdx.x;
        seqO[grow * MAX_LEN + (MAX_LEN - 1)] = 0.0f;
        lpsO[grow * MAX_LEN + (MAX_LEN - 1)] = 0.0f;
        entO[grow * MAX_LEN + (MAX_LEN - 1)] = 0.0f;
    }
}

// ---------------------------------------------------------------------------
// Launcher
// ---------------------------------------------------------------------------
extern "C" void kernel_launch(void* const* d_in, const int* in_sizes, int n_in,
                              void* d_out, int out_size, void* d_ws, size_t ws_size,
                              hipStream_t stream) {
    (void)in_sizes; (void)n_in; (void)out_size; (void)ws_size;

    const float* x       = (const float*)d_in[0];
    const float* agent_w = (const float*)d_in[1];
    const float* agent_b = (const float*)d_in[2];
    const float* W_ih    = (const float*)d_in[3];
    const float* W_hh    = (const float*)d_in[4];
    const float* b_ih    = (const float*)d_in[5];
    const float* b_hh    = (const float*)d_in[6];
    const float* out_w   = (const float*)d_in[7];
    const float* out_b   = (const float*)d_in[8];
    const float* emb     = (const float*)d_in[9];
    const float* sos     = (const float*)d_in[10];

    unsigned char* ws = (unsigned char*)d_ws;
    u16* x_bf   = (u16*)(ws + X_BF_OFF);
    u16* agw_bf = (u16*)(ws + AGW_OFF);
    u16* wih_bf = (u16*)(ws + WIH_OFF);
    u16* whh_bf = (u16*)(ws + WHH_OFF);
    u16* outw_bf= (u16*)(ws + OUTW_OFF);
    u16* emb_bf = (u16*)(ws + EMB_OFF);

    f2bf_kernel<<<512, 256, 0, stream>>>(x,       x_bf,    B_TOT * IN_DIM);
    f2bf_kernel<<<128, 256, 0, stream>>>(agent_w, agw_bf,  H_DIM * IN_DIM);
    f2bf_kernel<<<128, 256, 0, stream>>>(W_ih,    wih_bf,  3 * H_DIM * E_DIM);
    f2bf_kernel<<<128, 256, 0, stream>>>(W_hh,    whh_bf,  3 * H_DIM * H_DIM);
    f2bf_kernel<<<128, 256, 0, stream>>>(out_w,   outw_bf, V_DIM * H_DIM);
    f2bf_kernel<<<128, 256, 0, stream>>>(emb,     emb_bf,  V_DIM * E_DIM);

    rnn_sender_kernel<<<NBLOCKS, NTHREADS, LDS_BYTES, stream>>>(
        x_bf, agw_bf, agent_b, wih_bf, whh_bf, b_ih, b_hh,
        outw_bf, out_b, emb_bf, sos, (float*)d_out);
}
